// TransformerBlock_7773890806156
// MI455X (gfx1250) — matrix operand order repaired and checked by reading, not compiled, not run on hardware
//
#include <hip/hip_runtime.h>

typedef unsigned short u16;
typedef __bf16 bf16_t;
typedef bf16_t v16bf __attribute__((ext_vector_type(16)));
typedef float  v8f   __attribute__((ext_vector_type(8)));
typedef unsigned int u32x4 __attribute__((ext_vector_type(4)));
typedef int i32x8 __attribute__((ext_vector_type(8)));
typedef int i32x4 __attribute__((ext_vector_type(4)));

#define B_   2
#define N_   2048
#define D_   1024
#define H_   16
#define DPH_ 64
#define MTOT (B_ * N_)
#define NEGV (-1e9f)
#define EPS_ 1e-5f

#define HAS_TDM (__has_builtin(__builtin_amdgcn_tensor_load_to_lds) && \
                 __has_builtin(__builtin_amdgcn_s_wait_tensorcnt))

union FragU { v16bf v; uint4 q[2]; };

__device__ inline u16 f2bf(float f) {
  unsigned u = __float_as_uint(f);
  unsigned r = u + 0x7fffu + ((u >> 16) & 1u);
  return (u16)(r >> 16);
}

// A fragment 16x32 (MxK), source row-major [row][k], ld in elements.
__device__ inline v16bf load_a_frag(const u16* base, int ld, int lane) {
  int r = lane & 15, hi = lane >> 4;
  const u16* p = base + (size_t)r * ld + hi * 16;
  FragU f;
  f.q[0] = *reinterpret_cast<const uint4*>(p);
  f.q[1] = *reinterpret_cast<const uint4*>(p + 8);
  return f.v;
}

// B fragment 32x16 (KxN), source stored as B^T row-major [col][k], ld in elems.
__device__ inline v16bf load_b_frag(const u16* base, int ld, int lane) {
  int c = lane & 15, hi = lane >> 4;
  const u16* p = base + (size_t)c * ld + hi * 16;
  FragU f;
  f.q[0] = *reinterpret_cast<const uint4*>(p);
  f.q[1] = *reinterpret_cast<const uint4*>(p + 8);
  return f.v;
}

// B fragment from an LDS-staged [64][32] u16 tile (rows = output columns).
__device__ inline v16bf load_b_frag_lds(const u16* tile, int row0, int lane) {
  int c = lane & 15, hi = lane >> 4;
  const u16* p = tile + (size_t)(row0 + c) * 32 + hi * 16;
  FragU f;
  f.q[0] = *reinterpret_cast<const uint4*>(p);
  f.q[1] = *reinterpret_cast<const uint4*>(p + 8);
  return f.v;
}

__device__ inline v8f wmma_bf16(v16bf a, v16bf b, v8f c) {
  return __builtin_amdgcn_wmma_f32_16x16x32_bf16(false, a, false, b, (short)0, c,
                                                 false, false);
}

// ---------------------------------------------------------------------------
// TDM staging of a 64-row x 32-elem bf16 tile (rows strided by ld_elems) into
// LDS. Descriptor per CDNA5 ISA. 6-arg builtin form
// (clang-23 / therock-10.0). Fallback: cooperative synchronous copy.
// ---------------------------------------------------------------------------
__device__ inline void stage_b_tile(u16* lds_dst, const u16* gsrc, int ld_elems) {
#if HAS_TDM
  if (threadIdx.x < 32) {
    unsigned long long ga = (unsigned long long)(uintptr_t)gsrc;
    unsigned lds_off = (unsigned)(uintptr_t)lds_dst;  // low 32 bits = LDS offset
    u32x4 g0;
    g0[0] = 1u;                                        // count=1, user mode
    g0[1] = lds_off;                                   // lds_addr
    g0[2] = (unsigned)(ga & 0xffffffffu);              // global_addr[31:0]
    g0[3] = (unsigned)((ga >> 32) & 0x01ffffffu) | (2u << 30);  // addr[56:32]|type=2
    unsigned td0 = (unsigned)ld_elems;   // tensor_dim0 = full row length
    unsigned td1 = 0x00100000u;          // tensor_dim1: large, no OOB clipping
    unsigned st0 = (unsigned)ld_elems;   // tensor_dim0_stride
    i32x8 g1;
    g1[0] = (int)(1u << 16);                           // data_size=1 (2 bytes)
    g1[1] = (int)((td0 & 0xffffu) << 16);              // [31:16]=dim0[15:0]
    g1[2] = (int)(((td0 >> 16) & 0xffffu) | ((td1 & 0xffffu) << 16));
    g1[3] = (int)(((td1 >> 16) & 0xffffu) | (32u << 16));   // tile_dim0=32
    g1[4] = (int)(64u);                                // tile_dim1=64, tile_dim2=0
    g1[5] = (int)st0;                                  // dim0_stride[31:0]
    g1[6] = 0;                                         // stride[47:32]=0, d1stride lo=0
    g1[7] = 0;                                         // d1stride hi = 0
    i32x4 g2 = {0, 0, 0, 0};
    i32x4 g3 = {0, 0, 0, 0};
    i32x8 g4 = {0, 0, 0, 0, 0, 0, 0, 0};
    __builtin_amdgcn_tensor_load_to_lds(g0, g1, g2, g3, g4, 0);
  }
#else
  int t = threadIdx.x;
#pragma unroll
  for (int j = t; j < 256; j += 128) {   // 256 x uint4 = 4KB tile
    int row = j >> 2, ch = j & 3;
    reinterpret_cast<uint4*>(lds_dst)[j] =
        *reinterpret_cast<const uint4*>(gsrc + (size_t)row * ld_elems + ch * 8);
  }
#endif
}

__device__ inline void stage_wait() {
#if HAS_TDM
  if (threadIdx.x < 32) __builtin_amdgcn_s_wait_tensorcnt(0);
#endif
}

__global__ void cvt_bf16_kernel(const float4* __restrict__ src,
                                uint2* __restrict__ dst, int n4) {
  int i = blockIdx.x * blockDim.x + threadIdx.x;
  if (i < n4) {
    float4 f = src[i];
    uint2 o;
    o.x = (unsigned)f2bf(f.x) | ((unsigned)f2bf(f.y) << 16);
    o.y = (unsigned)f2bf(f.z) | ((unsigned)f2bf(f.w) << 16);
    dst[i] = o;
  }
}

// ---------------------------------------------------------------------------
// Shared GEMM core: block = 128 threads (4 waves), computes 128 rows x 64 cols.
// Each wave: 32 rows x 64 cols (8 accumulators). B tile staged in LDS by TDM,
// double-buffered; A fragments double-buffered in registers.
// ---------------------------------------------------------------------------
#define GEMM_CORE(Asrc, Bsrc, ACCS)                                            \
  __shared__ __align__(16) u16 Blds[2][64 * 32];                               \
  int lane = threadIdx.x & 31, wv = threadIdx.x >> 5;                          \
  int mbase = blockIdx.x * 128 + wv * 32;                                      \
  int nbase = blockIdx.y * 64;                                                 \
  v8f ACCS[8];                                                                 \
  _Pragma("unroll") for (int i = 0; i < 8; ++i) {                              \
    v8f z = {0, 0, 0, 0, 0, 0, 0, 0};                                          \
    ACCS[i] = z;                                                               \
  }                                                                            \
  v16bf a0 = load_a_frag(Asrc + (size_t)mbase * D_, D_, lane);                 \
  v16bf a1 = load_a_frag(Asrc + (size_t)(mbase + 16) * D_, D_, lane);          \
  stage_b_tile(Blds[0], Bsrc + (size_t)nbase * D_, D_);                        \
  for (int kk = 0; kk < D_; kk += 32) {                                        \
    int p = (kk >> 5) & 1;                                                     \
    stage_wait();                                                              \
    __syncthreads();                                                           \
    if (kk + 32 < D_)                                                          \
      stage_b_tile(Blds[p ^ 1], Bsrc + (size_t)nbase * D_ + kk + 32, D_);      \
    v16bf a0n = a0, a1n = a1;                                                  \
    if (kk + 32 < D_) {                                                        \
      a0n = load_a_frag(Asrc + (size_t)mbase * D_ + kk + 32, D_, lane);        \
      a1n = load_a_frag(Asrc + (size_t)(mbase + 16) * D_ + kk + 32, D_, lane); \
    }                                                                          \
    _Pragma("unroll") for (int jt = 0; jt < 4; ++jt) {                         \
      v16bf bfr = load_b_frag_lds(Blds[p], jt * 16, lane);                     \
      ACCS[jt] = wmma_bf16(a0, bfr, ACCS[jt]);                                 \
      ACCS[4 + jt] = wmma_bf16(a1, bfr, ACCS[4 + jt]);                         \
    }                                                                          \
    a0 = a0n;                                                                  \
    a1 = a1n;                                                                  \
  }

// ---------------------------------------------------------------------------
// QKV projection. grid = (MTOT/128, D/64, 3), block = 128
// ---------------------------------------------------------------------------
__global__ __launch_bounds__(128)
void qkv_gemm_kernel(const u16* __restrict__ Xb,
                     const u16* __restrict__ Wqb, const u16* __restrict__ Wkb,
                     const u16* __restrict__ Wvb,
                     const float* __restrict__ bq, const float* __restrict__ bk,
                     const float* __restrict__ bv,
                     u16* __restrict__ Qh, u16* __restrict__ Kh,
                     u16* __restrict__ Vt) {
  int which = blockIdx.z;
  const u16* Wb = (which == 0) ? Wqb : (which == 1) ? Wkb : Wvb;
  const float* bias = (which == 0) ? bq : (which == 1) ? bk : bv;

  GEMM_CORE(Xb, Wb, acc)

  int c = lane & 15, hi = lane >> 4;
#pragma unroll
  for (int half = 0; half < 2; ++half) {
#pragma unroll
    for (int jt = 0; jt < 4; ++jt) {
      v8f a = acc[half * 4 + jt];
      int col = nbase + jt * 16 + c;
      int h = col >> 6, dph = col & 63;
      float bval = bias[col];
#pragma unroll
      for (int g = 0; g < 8; ++g) {
        int m = mbase + half * 16 + g + hi * 8;
        int b = m >> 11, n = m & (N_ - 1);
        float val = a[g] + bval;
        if (which == 0) {
          Qh[(((size_t)(b * H_ + h)) * N_ + n) * DPH_ + dph] = f2bf(val * 0.125f);
        } else if (which == 1) {
          Kh[(((size_t)(b * H_ + h)) * N_ + n) * DPH_ + dph] = f2bf(val);
        } else {
          Vt[(((size_t)(b * H_ + h)) * DPH_ + dph) * N_ + n] = f2bf(val);
        }
      }
    }
  }
}

// ---------------------------------------------------------------------------
// Flash attention. grid = (N/64, H, B), block = 128 (each wave: 16 q-rows).
// ---------------------------------------------------------------------------
__global__ __launch_bounds__(128)
void flash_attn_kernel(const u16* __restrict__ Qh, const u16* __restrict__ Kh,
                       const u16* __restrict__ Vt, const int* __restrict__ pad,
                       float* __restrict__ ctx) {
  __shared__ __align__(64) u16 Plds[4][16][32];
  int lane = threadIdx.x & 31, wv = threadIdx.x >> 5;
  int qb = blockIdx.x, h = blockIdx.y, b = blockIdx.z;
  int qrow0 = qb * 64 + wv * 16;
  const u16* Qhead = Qh + ((size_t)(b * H_ + h)) * N_ * DPH_;
  const u16* Khead = Kh + ((size_t)(b * H_ + h)) * N_ * DPH_;
  const u16* Vhead = Vt + ((size_t)(b * H_ + h)) * DPH_ * N_;
  const int* padb = pad + (size_t)b * N_;
  int c = lane & 15, hi = lane >> 4;

  v16bf qa0 = load_a_frag(Qhead + (size_t)qrow0 * DPH_ + 0, DPH_, lane);
  v16bf qa1 = load_a_frag(Qhead + (size_t)qrow0 * DPH_ + 32, DPH_, lane);

  v8f o0 = {0,0,0,0,0,0,0,0};
  v8f o1 = o0, o2 = o0, o3 = o0;
  float mrow[8], lrow[8];
#pragma unroll
  for (int g = 0; g < 8; ++g) { mrow[g] = -1e30f; lrow[g] = 0.0f; }

  int kend = ((qrow0 + 16) + 31) & ~31;
  if (kend > N_) kend = N_;

  for (int kk = 0; kk < kend; kk += 32) {
    if (kk + 32 < kend) {
      __builtin_prefetch(Khead + (size_t)(kk + 32 + c) * DPH_, 0, 1);
      __builtin_prefetch(Vhead + (size_t)c * N_ + kk + 32, 0, 1);
    }
    // Preload ALL fragments for this chunk so loads overlap WMMAs + softmax.
    v16bf kb00 = load_b_frag(Khead + (size_t)kk * DPH_ + 0, DPH_, lane);
    v16bf kb01 = load_b_frag(Khead + (size_t)kk * DPH_ + 32, DPH_, lane);
    v16bf kb10 = load_b_frag(Khead + (size_t)(kk + 16) * DPH_ + 0, DPH_, lane);
    v16bf kb11 = load_b_frag(Khead + (size_t)(kk + 16) * DPH_ + 32, DPH_, lane);
    v16bf vb0 = load_b_frag(Vhead + (size_t)0 * N_ + kk, N_, lane);
    v16bf vb1 = load_b_frag(Vhead + (size_t)16 * N_ + kk, N_, lane);
    v16bf vb2 = load_b_frag(Vhead + (size_t)32 * N_ + kk, N_, lane);
    v16bf vb3 = load_b_frag(Vhead + (size_t)48 * N_ + kk, N_, lane);
    int pm0 = padb[kk + c];
    int pm1 = padb[kk + 16 + c];

    v8f s0 = {0,0,0,0,0,0,0,0};
    v8f s1 = s0;
    s0 = wmma_bf16(qa0, kb00, s0);
    s0 = wmma_bf16(qa1, kb01, s0);
    s1 = wmma_bf16(qa0, kb10, s1);
    s1 = wmma_bf16(qa1, kb11, s1);

#pragma unroll
    for (int g = 0; g < 8; ++g) {
      int qr = qrow0 + g + hi * 8;
      float a0 = ((kk + c) <= qr && pm0 == 0) ? s0[g] : NEGV;
      float a1 = ((kk + 16 + c) <= qr && pm1 == 0) ? s1[g] : NEGV;
      float cand = fmaxf(a0, a1);
      cand = fmaxf(cand, __shfl_xor(cand, 1, 32));
      cand = fmaxf(cand, __shfl_xor(cand, 2, 32));
      cand = fmaxf(cand, __shfl_xor(cand, 4, 32));
      cand = fmaxf(cand, __shfl_xor(cand, 8, 32));
      float mnew = fmaxf(mrow[g], cand);
      float alpha = __expf(mrow[g] - mnew);
      float p0 = __expf(a0 - mnew);
      float p1 = __expf(a1 - mnew);
      float ls = p0 + p1;
      ls += __shfl_xor(ls, 1, 32);
      ls += __shfl_xor(ls, 2, 32);
      ls += __shfl_xor(ls, 4, 32);
      ls += __shfl_xor(ls, 8, 32);
      lrow[g] = lrow[g] * alpha + ls;
      mrow[g] = mnew;
      o0[g] *= alpha; o1[g] *= alpha; o2[g] *= alpha; o3[g] *= alpha;
      Plds[wv][g + hi * 8][c] = f2bf(p0);
      Plds[wv][g + hi * 8][c + 16] = f2bf(p1);
    }

    // Re-read P in A-fragment layout (wave-private region; same-wave DS order)
    FragU pf;
    const u16* pp = &Plds[wv][c][hi * 16];
    pf.q[0] = *reinterpret_cast<const uint4*>(pp);
    pf.q[1] = *reinterpret_cast<const uint4*>(pp + 8);
    v16bf pa = pf.v;

    o0 = wmma_bf16(pa, vb0, o0);
    o1 = wmma_bf16(pa, vb1, o1);
    o2 = wmma_bf16(pa, vb2, o2);
    o3 = wmma_bf16(pa, vb3, o3);
  }

#pragma unroll
  for (int g = 0; g < 8; ++g) {
    float inv = 1.0f / lrow[g];
    int n = qrow0 + g + hi * 8;
    size_t rowoff = ((size_t)b * N_ + n) * D_ + (size_t)h * DPH_;
    ctx[rowoff + 0 + c]  = o0[g] * inv;
    ctx[rowoff + 16 + c] = o1[g] * inv;
    ctx[rowoff + 32 + c] = o2[g] * inv;
    ctx[rowoff + 48 + c] = o3[g] * inv;
  }
}

// ---------------------------------------------------------------------------
// Fused (optional add) + LayerNorm. grid = MTOT, block = 256 (8 waves).
// ---------------------------------------------------------------------------
__device__ inline float blockReduceSum(float v) {
#pragma unroll
  for (int o = 16; o > 0; o >>= 1) v += __shfl_xor(v, o, 32);
  __shared__ float sh[8];
  int lane = threadIdx.x & 31, w = threadIdx.x >> 5;
  if (lane == 0) sh[w] = v;
  __syncthreads();
  float r = 0.0f;
#pragma unroll
  for (int i = 0; i < 8; ++i) r += sh[i];
  __syncthreads();
  return r;
}

__global__ __launch_bounds__(256)
void add_ln_kernel(const float* __restrict__ xa, const float* __restrict__ xb,
                   const float* __restrict__ gam, const float* __restrict__ bet,
                   float* __restrict__ outf, u16* __restrict__ outb) {
  int row = blockIdx.x;
  const float* pa = xa + (size_t)row * D_;
  const float* pb = xb ? xb + (size_t)row * D_ : nullptr;
  int t = threadIdx.x;
  float v[4];
  float s = 0.0f;
#pragma unroll
  for (int i = 0; i < 4; ++i) {
    int col = t + i * 256;
    float x = pa[col];
    if (pb) x += pb[col];
    v[i] = x;
    s += x;
  }
  float mean = blockReduceSum(s) * (1.0f / D_);
  float vs = 0.0f;
#pragma unroll
  for (int i = 0; i < 4; ++i) {
    float d = v[i] - mean;
    vs += d * d;
  }
  float var = blockReduceSum(vs) * (1.0f / D_);
  float rs = rsqrtf(var + EPS_);
#pragma unroll
  for (int i = 0; i < 4; ++i) {
    int col = t + i * 256;
    float y = (v[i] - mean) * rs * gam[col] + bet[col];
    outf[(size_t)row * D_ + col] = y;
    if (outb) outb[(size_t)row * D_ + col] = f2bf(y);
  }
}

// ---------------------------------------------------------------------------
// FFN: sum2 = attn_out + relu(attn_out @ Wo^T + bo). grid = (MTOT/128, D/64)
// ---------------------------------------------------------------------------
__global__ __launch_bounds__(128)
void ffn_gemm_kernel(const u16* __restrict__ Ab, const u16* __restrict__ Wob,
                     const float* __restrict__ bo,
                     const float* __restrict__ attn_out,
                     float* __restrict__ sum2) {
  GEMM_CORE(Ab, Wob, acc)

  int c = lane & 15, hi = lane >> 4;
#pragma unroll
  for (int half = 0; half < 2; ++half) {
#pragma unroll
    for (int jt = 0; jt < 4; ++jt) {
      v8f a = acc[half * 4 + jt];
      int col = nbase + jt * 16 + c;
      float bval = bo[col];
#pragma unroll
      for (int g = 0; g < 8; ++g) {
        int m = mbase + half * 16 + g + hi * 8;
        float r = fmaxf(a[g] + bval, 0.0f);
        size_t idx = (size_t)m * D_ + col;
        sum2[idx] = attn_out[idx] + r;
      }
    }
  }
}

// ---------------------------------------------------------------------------
extern "C" void kernel_launch(void* const* d_in, const int* in_sizes, int n_in,
                              void* d_out, int out_size, void* d_ws,
                              size_t ws_size, hipStream_t stream) {
  (void)in_sizes; (void)n_in; (void)out_size; (void)ws_size;
  const float* x   = (const float*)d_in[0];
  const int*   pad = (const int*)d_in[1];
  const float* Wq  = (const float*)d_in[2];
  const float* bq  = (const float*)d_in[3];
  const float* Wk  = (const float*)d_in[4];
  const float* bk  = (const float*)d_in[5];
  const float* Wv  = (const float*)d_in[6];
  const float* bv  = (const float*)d_in[7];
  const float* Wo  = (const float*)d_in[8];
  const float* bo  = (const float*)d_in[9];
  const float* lng = (const float*)d_in[10];
  const float* lnb = (const float*)d_in[11];
  float* out = (float*)d_out;

  char* p = (char*)d_ws;
  auto alloc = [&](size_t bytes) -> void* {
    void* r = (void*)p;
    p += (bytes + 255) & ~(size_t)255;
    return r;
  };
  u16* Xb      = (u16*)alloc((size_t)MTOT * D_ * 2);
  u16* Wqb     = (u16*)alloc((size_t)D_ * D_ * 2);
  u16* Wkb     = (u16*)alloc((size_t)D_ * D_ * 2);
  u16* Wvb     = (u16*)alloc((size_t)D_ * D_ * 2);
  u16* Wob     = (u16*)alloc((size_t)D_ * D_ * 2);
  u16* Qh      = (u16*)alloc((size_t)B_ * H_ * N_ * DPH_ * 2);
  u16* Kh      = (u16*)alloc((size_t)B_ * H_ * N_ * DPH_ * 2);
  u16* Vt      = (u16*)alloc((size_t)B_ * H_ * DPH_ * N_ * 2);
  float* ctx   = (float*)alloc((size_t)MTOT * D_ * 4);
  float* attnf = (float*)alloc((size_t)MTOT * D_ * 4);
  u16* attnb   = (u16*)alloc((size_t)MTOT * D_ * 2);
  float* sum2  = (float*)alloc((size_t)MTOT * D_ * 4);

  const int nx4 = MTOT * D_ / 4;
  const int nw4 = D_ * D_ / 4;
  cvt_bf16_kernel<<<dim3((nx4 + 255) / 256), dim3(256), 0, stream>>>(
      (const float4*)x, (uint2*)Xb, nx4);
  cvt_bf16_kernel<<<dim3((nw4 + 255) / 256), dim3(256), 0, stream>>>(
      (const float4*)Wq, (uint2*)Wqb, nw4);
  cvt_bf16_kernel<<<dim3((nw4 + 255) / 256), dim3(256), 0, stream>>>(
      (const float4*)Wk, (uint2*)Wkb, nw4);
  cvt_bf16_kernel<<<dim3((nw4 + 255) / 256), dim3(256), 0, stream>>>(
      (const float4*)Wv, (uint2*)Wvb, nw4);
  cvt_bf16_kernel<<<dim3((nw4 + 255) / 256), dim3(256), 0, stream>>>(
      (const float4*)Wo, (uint2*)Wob, nw4);

  qkv_gemm_kernel<<<dim3(MTOT / 128, D_ / 64, 3), dim3(128), 0, stream>>>(
      Xb, Wqb, Wkb, Wvb, bq, bk, bv, Qh, Kh, Vt);

  flash_attn_kernel<<<dim3(N_ / 64, H_, B_), dim3(128), 0, stream>>>(
      Qh, Kh, Vt, pad, ctx);

  add_ln_kernel<<<dim3(MTOT), dim3(256), 0, stream>>>(x, ctx, lng, lnb, attnf,
                                                      attnb);

  ffn_gemm_kernel<<<dim3(MTOT / 128, D_ / 64, 1), dim3(128), 0, stream>>>(
      attnb, Wob, bo, attnf, sum2);

  add_ln_kernel<<<dim3(MTOT), dim3(256), 0, stream>>>(sum2, nullptr, lng, lnb,
                                                      out, nullptr);
}
